// HolidayAwareAttention_24988119728244
// MI455X (gfx1250) — compile-verified, matrix operand order checked by reading, not run on hardware
//
#include <hip/hip_runtime.h>
#include <hip/hip_bf16.h>

// ---------------------------------------------------------------------------
// Transformer block for MI455X (gfx1250, wave32, WMMA).
// bf16 activations, f32 accumulation via v_wmma_f32_16x16x32_bf16.
// A-tiles staged to LDS with global_load_async_to_lds_b128 (ASYNCcnt).
// ---------------------------------------------------------------------------

typedef __bf16 bf16;
typedef __attribute__((ext_vector_type(16))) __bf16 v16bf;
typedef __attribute__((ext_vector_type(8)))  float  v8f;

#define BB   4
#define TT   4096
#define CC   256
#define HH   4
#define DD   64
#define MM   (BB * TT)      // 16384 tokens
#define N3   (3 * CC)       // 768
#define LNEPS 1e-5f
#define BSTR 34             // transposed-B LDS stride (bank-spread, even)

// K index inside a 16x32 (16-bit) A/B fragment for VGPR r, lane-half hi.
// ISA 7.12.2: VGPR0..3 lanes0-15 hold K=0..7 (pairs), lanes16-31 K=8..15; VGPR4..7 -> +16.
__device__ __forceinline__ int kbase(int r, int hi) {
  return ((r & 4) << 2) | (hi << 3) | ((r & 3) << 1);
}

union FragAB {
  v16bf v;
  unsigned int u[8];
  bf16 b[16];
};

__device__ __forceinline__ unsigned lds_off32(const void* p) {
  return (unsigned)(unsigned long long)(uintptr_t)p;   // low 32 bits = LDS byte offset
}

// ---------------------------------------------------------------------------
// fp32 -> bf16 convert
// ---------------------------------------------------------------------------
__global__ __launch_bounds__(256) void cvt_bf16_kernel(const float* __restrict__ in,
                                                       bf16* __restrict__ out, int n) {
  int i = blockIdx.x * 256 + threadIdx.x;
  if (i < n) out[i] = (bf16)in[i];
}

// ---------------------------------------------------------------------------
// Tiled GEMM: Out = A[M x K] * W[K x N] (+bias, + epilogue)
// Block tile 256x32 (8 waves, each a 32x32 tile = 4 WMMA per K=32 chunk).
// A tile: async-DMA to LDS (row-major, stride 32).
// B tile: stored transposed [n][k] in LDS so fragments load as b32 pairs.
// mode 0: QKV scatter -> q/k bf16 [B,H,T,D], v bf16 [B,H,D,T]
// mode 1: + residual  -> f32 out [M x N]
// mode 2: GELU        -> bf16 out [M x N]
// ---------------------------------------------------------------------------
__global__ __launch_bounds__(256) void gemm_bf16_kernel(
    const bf16* __restrict__ A, const bf16* __restrict__ W,
    const float* __restrict__ bias, int K, int N, int mode,
    const float* __restrict__ res, float* __restrict__ outf,
    bf16* __restrict__ outb, bf16* __restrict__ oq,
    bf16* __restrict__ okk, bf16* __restrict__ ov) {
  __shared__ __align__(16) bf16 sA[256 * 32];
  __shared__ __align__(16) bf16 sBt[32 * BSTR];

  const int tid  = threadIdx.x;
  const int lane = tid & 31;
  const int wave = tid >> 5;           // 0..7, owns rows [wave*32, wave*32+32)
  const int hi   = (lane >> 4) & 1;
  const int m0   = blockIdx.y * 256;
  const int n0   = blockIdx.x * 32;

  const int arow = tid >> 2, acol = (tid & 3) << 3;  // + i*64 rows, 8 bf16 each
  const int brow = tid >> 3, bcol = (tid & 7) << 2;  // 32 k-rows x 4 n each

  v8f acc[2][2];
#pragma unroll
  for (int ms = 0; ms < 2; ++ms)
#pragma unroll
    for (int ns = 0; ns < 2; ++ns)
      acc[ms][ns] = (v8f){0.f, 0.f, 0.f, 0.f, 0.f, 0.f, 0.f, 0.f};

  for (int k0 = 0; k0 < K; k0 += 32) {
    __syncthreads();
    // ---- A tile: async DMA, 4 x 16B per lane (rows arow + i*64) ----
#pragma unroll
    for (int i = 0; i < 4; ++i) {
      unsigned lo = lds_off32(sA + (arow + i * 64) * 32 + acol);
      unsigned long long ga =
          (unsigned long long)(uintptr_t)(A + (size_t)(m0 + arow + i * 64) * K + k0 + acol);
      asm volatile("global_load_async_to_lds_b128 %0, %1, off"
                   :: "v"(lo), "v"(ga) : "memory");
    }
    // ---- B tile: load [k][n] row-major, store transposed [n][k] ----
    {
      uint2 w2 = *reinterpret_cast<const uint2*>(W + (size_t)(k0 + brow) * N + n0 + bcol);
      bf16 wb[4];
      *reinterpret_cast<uint2*>(wb) = w2;
#pragma unroll
      for (int i = 0; i < 4; ++i) sBt[(bcol + i) * BSTR + brow] = wb[i];
    }
    if (k0 + 32 < K)
      __builtin_prefetch(W + (size_t)(k0 + 32 + brow) * N + n0 + bcol, 0, 1);

    asm volatile("s_wait_asynccnt 0" ::: "memory");
    __syncthreads();

    // ---- fragments: all b32 LDS loads ----
    FragAB fa[2], fb[2];
#pragma unroll
    for (int ms = 0; ms < 2; ++ms) {
      const bf16* ap = sA + (wave * 32 + ms * 16 + (lane & 15)) * 32;
#pragma unroll
      for (int r = 0; r < 8; ++r)
        fa[ms].u[r] = *reinterpret_cast<const unsigned int*>(ap + kbase(r, hi));
    }
#pragma unroll
    for (int ns = 0; ns < 2; ++ns) {
      const bf16* bp = sBt + (ns * 16 + (lane & 15)) * BSTR;
#pragma unroll
      for (int r = 0; r < 8; ++r)
        fb[ns].u[r] = *reinterpret_cast<const unsigned int*>(bp + kbase(r, hi));
    }
#pragma unroll
    for (int ms = 0; ms < 2; ++ms)
#pragma unroll
      for (int ns = 0; ns < 2; ++ns)
        acc[ms][ns] = __builtin_amdgcn_wmma_f32_16x16x32_bf16(
            false, fa[ms].v, false, fb[ns].v, (short)0, acc[ms][ns], false, false);
  }

  // ---- epilogue ----
#pragma unroll
  for (int ns = 0; ns < 2; ++ns) {
    const int gn = n0 + ns * 16 + (lane & 15);
    const float bv = bias[gn];
#pragma unroll
    for (int ms = 0; ms < 2; ++ms) {
#pragma unroll
      for (int r = 0; r < 8; ++r) {
        const int gm = m0 + wave * 32 + ms * 16 + r + (hi << 3);
        float val = acc[ms][ns][r] + bv;
        if (mode == 0) {               // QKV scatter
          int part = gn >> 8, cc = gn & 255;
          int hh = cc >> 6, dd = cc & 63;
          int bb = gm >> 12, ttok = gm & (TT - 1);
          if (part == 2) {             // V transposed: [B,H,D,T]
            ov[(((size_t)(bb * HH + hh)) * DD + dd) * TT + ttok] = (bf16)val;
          } else {
            bf16* dst = (part == 0) ? oq : okk;
            dst[(((size_t)(bb * HH + hh)) * TT + ttok) * DD + dd] = (bf16)val;
          }
        } else if (mode == 1) {        // bias + residual -> f32
          size_t idx = (size_t)gm * N + gn;
          outf[idx] = val + res[idx];
        } else {                       // exact GELU -> bf16
          float g = 0.5f * val * (1.0f + erff(val * 0.70710678118f));
          outb[(size_t)gm * N + gn] = (bf16)g;
        }
      }
    }
  }
}

// ---------------------------------------------------------------------------
// Flash attention: each wave owns a 16-row Q tile of one (b,h); streams keys
// in chunks of 32. 8 WMMAs per chunk. V is [B,H,D,T] so V-frags are b32 loads.
// ---------------------------------------------------------------------------
__global__ __launch_bounds__(256) void attn_kernel(const bf16* __restrict__ q,
                                                   const bf16* __restrict__ k,
                                                   const bf16* __restrict__ v,
                                                   bf16* __restrict__ o) {
  __shared__ __align__(16) bf16 sP[8 * 16 * 32];   // per-wave P staging

  const int tid  = threadIdx.x;
  const int lane = tid & 31;
  const int wave = tid >> 5;
  const int hi   = (lane >> 4) & 1;
  const int tile = blockIdx.x * 8 + wave;          // 0 .. B*H*(T/16)-1
  const int ntq  = TT / 16;                        // 256
  const int bh   = tile / ntq;
  const int tq   = tile - bh * ntq;
  const size_t rowbase = (size_t)bh * TT;

  bf16* myP = sP + wave * 512;

  // Q fragments (d = 0..31 and 32..63), loaded once.
  FragAB qf0, qf1;
  {
    const bf16* qp = q + (rowbase + tq * 16 + (lane & 15)) * DD;
#pragma unroll
    for (int r = 0; r < 8; ++r) {
      qf0.u[r] = *reinterpret_cast<const unsigned int*>(qp + kbase(r, hi));
      qf1.u[r] = *reinterpret_cast<const unsigned int*>(qp + 32 + kbase(r, hi));
    }
  }

  v8f oacc[4];
#pragma unroll
  for (int j = 0; j < 4; ++j) oacc[j] = (v8f){0.f, 0.f, 0.f, 0.f, 0.f, 0.f, 0.f, 0.f};
  float rowm[8], rows[8];
#pragma unroll
  for (int r = 0; r < 8; ++r) { rowm[r] = -3.0e38f; rows[r] = 0.f; }

  const float scale = 0.125f;  // D^-0.5, D=64

  for (int s0 = 0; s0 < TT; s0 += 32) {
    // ---- S = Q * K^T for 32 keys (two 16x16 N-tiles) ----
    v8f sacc[2];
#pragma unroll
    for (int nt = 0; nt < 2; ++nt) {
      sacc[nt] = (v8f){0.f, 0.f, 0.f, 0.f, 0.f, 0.f, 0.f, 0.f};
      const bf16* kp = k + (rowbase + s0 + nt * 16 + (lane & 15)) * DD;
      FragAB kf0, kf1;
#pragma unroll
      for (int r = 0; r < 8; ++r) {
        kf0.u[r] = *reinterpret_cast<const unsigned int*>(kp + kbase(r, hi));
        kf1.u[r] = *reinterpret_cast<const unsigned int*>(kp + 32 + kbase(r, hi));
      }
      sacc[nt] = __builtin_amdgcn_wmma_f32_16x16x32_bf16(false, qf0.v, false, kf0.v,
                                                         (short)0, sacc[nt], false, false);
      sacc[nt] = __builtin_amdgcn_wmma_f32_16x16x32_bf16(false, qf1.v, false, kf1.v,
                                                         (short)0, sacc[nt], false, false);
    }

    // ---- online softmax over this 32-col chunk (rows live in 16-lane halves) ----
    float corr[8];
#pragma unroll
    for (int r = 0; r < 8; ++r) {
      float c0 = sacc[0][r] * scale;
      float c1 = sacc[1][r] * scale;
      float mc = fmaxf(c0, c1);
#pragma unroll
      for (int off = 1; off <= 8; off <<= 1) mc = fmaxf(mc, __shfl_xor(mc, off, 32));
      float mn = fmaxf(rowm[r], mc);
      corr[r] = __expf(rowm[r] - mn);
      rowm[r] = mn;
      float p0 = __expf(c0 - mn);
      float p1 = __expf(c1 - mn);
      float rs = p0 + p1;
#pragma unroll
      for (int off = 1; off <= 8; off <<= 1) rs += __shfl_xor(rs, off, 32);
      rows[r] = rows[r] * corr[r] + rs;
      int rr = r + (hi << 3);
      myP[rr * 32 + (lane & 15)]      = (bf16)p0;
      myP[rr * 32 + 16 + (lane & 15)] = (bf16)p1;
    }
    asm volatile("s_wait_dscnt 0" ::: "memory");   // C-layout -> A-layout through LDS

    // ---- P fragment (16x32) ----
    FragAB pf;
    const bf16* pp = myP + (lane & 15) * 32;
#pragma unroll
    for (int r = 0; r < 8; ++r)
      pf.u[r] = *reinterpret_cast<const unsigned int*>(pp + kbase(r, hi));

    // ---- O = O*corr + P * V   (V-frags: contiguous b32 pairs along keys) ----
#pragma unroll
    for (int j = 0; j < 4; ++j) {
#pragma unroll
      for (int r = 0; r < 8; ++r) oacc[j][r] *= corr[r];
      FragAB vf;
      const bf16* vp = v + ((size_t)bh * DD + j * 16 + (lane & 15)) * TT + s0;
#pragma unroll
      for (int r = 0; r < 8; ++r)
        vf.u[r] = *reinterpret_cast<const unsigned int*>(vp + kbase(r, hi));
      oacc[j] = __builtin_amdgcn_wmma_f32_16x16x32_bf16(false, pf.v, false, vf.v,
                                                        (short)0, oacc[j], false, false);
    }
  }

  // ---- normalize + merge heads: write [B,T,C] bf16 ----
  const int b = bh >> 2, h = bh & 3;
#pragma unroll
  for (int r = 0; r < 8; ++r) {
    float inv = 1.0f / rows[r];
    int token = tq * 16 + r + (hi << 3);
#pragma unroll
    for (int j = 0; j < 4; ++j) {
      int col = h * DD + j * 16 + (lane & 15);
      o[((size_t)b * TT + token) * CC + col] = (bf16)(oacc[j][r] * inv);
    }
  }
}

// ---------------------------------------------------------------------------
// LayerNorm over C=256: one wave per token; butterfly reductions.
// ---------------------------------------------------------------------------
__global__ __launch_bounds__(256) void ln_kernel(const float* __restrict__ in,
                                                 const float* __restrict__ w,
                                                 const float* __restrict__ bias,
                                                 float* __restrict__ outf,
                                                 bf16* __restrict__ outb) {
  const int lane  = threadIdx.x & 31;
  const int token = blockIdx.x * 8 + (threadIdx.x >> 5);
  const float4* p = reinterpret_cast<const float4*>(in + (size_t)token * CC);
  float4 a0 = p[lane];
  float4 a1 = p[32 + lane];

  float sum = a0.x + a0.y + a0.z + a0.w + a1.x + a1.y + a1.z + a1.w;
  float sq  = a0.x * a0.x + a0.y * a0.y + a0.z * a0.z + a0.w * a0.w +
              a1.x * a1.x + a1.y * a1.y + a1.z * a1.z + a1.w * a1.w;
#pragma unroll
  for (int off = 1; off <= 16; off <<= 1) {
    sum += __shfl_xor(sum, off, 32);
    sq  += __shfl_xor(sq, off, 32);
  }
  const float mean = sum * (1.0f / CC);
  const float var  = sq * (1.0f / CC) - mean * mean;
  const float rstd = rsqrtf(var + LNEPS);

  float4* of = reinterpret_cast<float4*>(outf + (size_t)token * CC);
#pragma unroll
  for (int i = 0; i < 2; ++i) {
    float4 av = (i == 0) ? a0 : a1;
    int c = (i * 32 + lane) * 4;
    float4 y;
    y.x = w[c + 0] * (av.x - mean) * rstd + bias[c + 0];
    y.y = w[c + 1] * (av.y - mean) * rstd + bias[c + 1];
    y.z = w[c + 2] * (av.z - mean) * rstd + bias[c + 2];
    y.w = w[c + 3] * (av.w - mean) * rstd + bias[c + 3];
    of[i * 32 + lane] = y;
    if (outb) {
      bf16* ob = outb + (size_t)token * CC + c;
      ob[0] = (bf16)y.x; ob[1] = (bf16)y.y; ob[2] = (bf16)y.z; ob[3] = (bf16)y.w;
    }
  }
}

// ---------------------------------------------------------------------------
// Host launcher
// ---------------------------------------------------------------------------
extern "C" void kernel_launch(void* const* d_in, const int* in_sizes, int n_in,
                              void* d_out, int out_size, void* d_ws, size_t ws_size,
                              hipStream_t stream) {
  const float* x      = (const float*)d_in[0];
  const float* qkv_w  = (const float*)d_in[1];
  const float* qkv_b  = (const float*)d_in[2];
  const float* proj_w = (const float*)d_in[3];
  const float* proj_b = (const float*)d_in[4];
  const float* n1_w   = (const float*)d_in[5];
  const float* n1_b   = (const float*)d_in[6];
  const float* ffn_w1 = (const float*)d_in[7];
  const float* ffn_b1 = (const float*)d_in[8];
  const float* ffn_w2 = (const float*)d_in[9];
  const float* ffn_b2 = (const float*)d_in[10];
  const float* n2_w   = (const float*)d_in[11];
  const float* n2_b   = (const float*)d_in[12];
  float* out = (float*)d_out;

  char* ws = (char*)d_ws;
  size_t off = 0;
  auto take = [&](size_t bytes) { void* p = ws + off; off += (bytes + 255) & ~(size_t)255; return p; };

  bf16*  x_bf     = (bf16*)take((size_t)MM * CC * 2);
  bf16*  wqkv_bf  = (bf16*)take((size_t)CC * N3 * 2);
  bf16*  wproj_bf = (bf16*)take((size_t)CC * CC * 2);
  bf16*  w1_bf    = (bf16*)take((size_t)CC * N3 * 2);
  bf16*  w2_bf    = (bf16*)take((size_t)N3 * CC * 2);
  bf16*  q_bf     = (bf16*)take((size_t)MM * CC * 2);
  bf16*  k_bf     = (bf16*)take((size_t)MM * CC * 2);
  bf16*  v_bf     = (bf16*)take((size_t)MM * CC * 2);   // [B,H,D,T]
  bf16*  attn_bf  = (bf16*)take((size_t)MM * CC * 2);
  float* r1       = (float*)take((size_t)MM * CC * 4);
  float* x1f      = (float*)take((size_t)MM * CC * 4);
  bf16*  x1_bf    = (bf16*)take((size_t)MM * CC * 2);
  bf16*  h_bf     = (bf16*)take((size_t)MM * N3 * 2);
  float* r2       = (float*)take((size_t)MM * CC * 4);

  // 1) converts
  cvt_bf16_kernel<<<(MM * CC + 255) / 256, 256, 0, stream>>>(x, x_bf, MM * CC);
  cvt_bf16_kernel<<<(CC * N3 + 255) / 256, 256, 0, stream>>>(qkv_w, wqkv_bf, CC * N3);
  cvt_bf16_kernel<<<(CC * CC + 255) / 256, 256, 0, stream>>>(proj_w, wproj_bf, CC * CC);
  cvt_bf16_kernel<<<(CC * N3 + 255) / 256, 256, 0, stream>>>(ffn_w1, w1_bf, CC * N3);
  cvt_bf16_kernel<<<(N3 * CC + 255) / 256, 256, 0, stream>>>(ffn_w2, w2_bf, N3 * CC);

  // 2) QKV projection, scatter q/k -> [B,H,T,D], v -> [B,H,D,T]
  gemm_bf16_kernel<<<dim3(N3 / 32, MM / 256), 256, 0, stream>>>(
      x_bf, wqkv_bf, qkv_b, CC, N3, 0, nullptr, nullptr, nullptr, q_bf, k_bf, v_bf);

  // 3) flash attention -> [B,T,C] bf16
  attn_kernel<<<(BB * HH * (TT / 16)) / 8, 256, 0, stream>>>(q_bf, k_bf, v_bf, attn_bf);

  // 4) output projection + residual(x) -> r1 f32
  gemm_bf16_kernel<<<dim3(CC / 32, MM / 256), 256, 0, stream>>>(
      attn_bf, wproj_bf, proj_b, CC, CC, 1, x, r1, nullptr, nullptr, nullptr, nullptr);

  // 5) LN1 -> x1 (f32 + bf16)
  ln_kernel<<<MM / 8, 256, 0, stream>>>(r1, n1_w, n1_b, x1f, x1_bf);

  // 6) FFN1 + GELU -> h bf16
  gemm_bf16_kernel<<<dim3(N3 / 32, MM / 256), 256, 0, stream>>>(
      x1_bf, w1_bf, ffn_b1, CC, N3, 2, nullptr, nullptr, h_bf, nullptr, nullptr, nullptr);

  // 7) FFN2 + residual(x1) -> r2 f32
  gemm_bf16_kernel<<<dim3(CC / 32, MM / 256), 256, 0, stream>>>(
      h_bf, w2_bf, ffn_b2, N3, CC, 1, x1f, r2, nullptr, nullptr, nullptr, nullptr);

  // 8) LN2 -> output f32
  ln_kernel<<<MM / 8, 256, 0, stream>>>(r2, n2_w, n2_b, out, nullptr);
}
